// RITS_34067680592458
// MI455X (gfx1250) — matrix operand order repaired and checked by reading, not compiled, not run on hardware
//
#include <hip/hip_runtime.h>
#include <hip/hip_bf16.h>
#include <math.h>

typedef __attribute__((ext_vector_type(16))) _Float16 v16h;
typedef __attribute__((ext_vector_type(8)))  _Float16 v8h;
typedef __attribute__((ext_vector_type(8)))  float    v8f;

#define BB 512
#define TT 128
#define DD 128
#define HH 128
#define ROWS 16      // batch rows per workgroup (one WMMA M-tile)
#define NBLK (BB/ROWS)

#define WMMA_F16(acc, a, b) \
  acc = __builtin_amdgcn_wmma_f32_16x16x32_f16(false, (a), false, (b), (short)0, (acc), false, false)

// ---- fast transcendentals (native v_exp_f32 / v_rcp_f32) -------------------
__device__ __forceinline__ float fexp(float x)  { return __expf(x); }
__device__ __forceinline__ float frcp(float x)  { return __builtin_amdgcn_rcpf(x); }
__device__ __forceinline__ float fsig(float x)  { return frcp(1.0f + __expf(-x)); }
__device__ __forceinline__ float ftanh(float x) { return 1.0f - 2.0f * frcp(__expf(2.0f * x) + 1.0f); }

// ---- WMMA fragment loaders ------------------------------------------------
// A (16x32 f16, M=batch row): lanes 0-15 rows, element j<8 -> K=hi*8+j,
// j>=8 -> K=16+hi*8+(j-8).  Two aligned 16B LDS reads per lane.
__device__ __forceinline__ v16h load_a16(const _Float16* buf, int lane, int kbase, int ld) {
  const int r = lane & 15, hi = lane >> 4;
  const _Float16* p = buf + r * ld + kbase + hi * 8;
  v8h lo = *(const v8h*)(p);
  v8h up = *(const v8h*)(p + 16);
  v16h a;
#pragma unroll
  for (int i = 0; i < 8; ++i) { a[i] = lo[i]; a[8 + i] = up[i]; }
  return a;
}

// B (32x16 f16): lane -> N = n0 + (lane&15); element j<8 -> K = kbase + hi*8 + j, j>=8 -> K = kbase + 16 + hi*8 + (j-8).
// Weights stored [N][K] row-major f16: two 16B runs per lane, 32B apart (L2-resident).
__device__ __forceinline__ v16h load_b16(const _Float16* __restrict__ W, int lane,
                                         int n0, int kbase, int ldk) {
  const int n = n0 + (lane & 15), hi = lane >> 4;
  const _Float16* p = W + (size_t)n * ldk + kbase + hi * 8;
  v8h lo = *(const v8h*)(p);
  v8h up = *(const v8h*)(p + 16);
  v16h b;
#pragma unroll
  for (int i = 0; i < 8; ++i) { b[i] = lo[i]; b[8 + i] = up[i]; }
  return b;
}

// ---- prep kernels ---------------------------------------------------------
__global__ void cvt_f16(const float* __restrict__ s, _Float16* __restrict__ d, int n) {
  int i = blockIdx.x * 256 + threadIdx.x;
  if (i < n) d[i] = (_Float16)s[i];
}
// src [R][C] f32 row-major -> dst [C][R] f16 (dst[c*R+r] = src[r*C+c])
__global__ void cvt_f16_T(const float* __restrict__ s, _Float16* __restrict__ d, int R, int C) {
  int i = blockIdx.x * 256 + threadIdx.x;
  if (i < R * C) {
    int r = i / C, c = i - r * C;
    d[(size_t)c * R + r] = (_Float16)s[i];
  }
}
// denom[t] = sum over [B,D] of masks[:,t,:] + eps  (global coupling, done up-front)
__global__ void denom_kernel(const float* __restrict__ masks, float* __restrict__ denomT) {
  const int t = blockIdx.x;
  __shared__ float red[256];
  float s = 0.f;
  for (int i = threadIdx.x; i < BB * DD; i += 256) {
    int b = i >> 7, col = i & 127;
    s += masks[((size_t)b * TT + t) * DD + col];
  }
  red[threadIdx.x] = s;
  __syncthreads();
  for (int o = 128; o > 0; o >>= 1) {
    if (threadIdx.x < o) red[threadIdx.x] += red[threadIdx.x + o];
    __syncthreads();
  }
  if (threadIdx.x == 0) denomT[t] = red[0] + 1e-6f;
}

// ---- persistent recurrent kernel ------------------------------------------
__global__ __launch_bounds__(256)
void rits_persistent(const float* __restrict__ values, const float* __restrict__ masks,
                     const float* __restrict__ deltas,
                     const float* __restrict__ bth, const float* __restrict__ btx,
                     const float* __restrict__ bhr, const float* __restrict__ bfr,
                     const float* __restrict__ bwc, const float* __restrict__ bl,
                     const float* __restrict__ bd,  const float* __restrict__ Wo,
                     const float* __restrict__ bo,
                     const _Float16* __restrict__ WthF, const _Float16* __restrict__ WtxF,
                     const _Float16* __restrict__ WfrF, const _Float16* __restrict__ WhrT,
                     const _Float16* __restrict__ WwcT, const _Float16* __restrict__ WkT,
                     const _Float16* __restrict__ WrT,  const _Float16* __restrict__ WdT,
                     const float* __restrict__ denomT,
                     float* __restrict__ outPred, float* __restrict__ outImp,
                     float* __restrict__ outLoss) {
  __shared__ __align__(16) _Float16 hA[ROWS * HH];   // h state (f16 operand image)
  __shared__ __align__(16) _Float16 hB[ROWS * HH];   // decayed h for this step
  __shared__ __align__(16) float    c32[ROWS * HH];  // LSTM cell state (f32)
  __shared__ __align__(16) float    xhat32[ROWS * DD];
  __shared__ __align__(16) float    zhat32[ROWS * DD];
  __shared__ __align__(16) _Float16 xF[ROWS * DD];
  __shared__ __align__(16) _Float16 mF[ROWS * DD];
  __shared__ __align__(16) _Float16 dF[ROWS * DD];
  __shared__ __align__(16) _Float16 gxF[ROWS * DD];
  __shared__ __align__(16) _Float16 xcF[ROWS * DD];
  __shared__ __align__(16) _Float16 ccF[ROWS * DD];
  __shared__ float lossRow[ROWS];
  __shared__ float predRow[ROWS];

  const int tid  = threadIdx.x;
  const int lane = tid & 31;
  const int w    = tid >> 5;       // 8 waves; wave w owns N columns [w*16, w*16+16)
  const int n0   = w * 16;
  const int rb   = blockIdx.x * ROWS;
  const int rloc = (lane >> 4) * 8;          // C/D fragment: row = rloc + r
  const int cl   = lane & 15;                // C/D fragment: local column

  for (int i = tid; i < ROWS * HH; i += 256) { hA[i] = (_Float16)0.f; c32[i] = 0.f; }
  if (tid < ROWS) { lossRow[tid] = 0.f; predRow[tid] = 0.f; }
  __syncthreads();

  for (int t = 0; t < TT; ++t) {
    // Opaque zero offset: defeats LICM so weight fragments stream from the
    // L2-resident f16 image every step instead of being hoisted + spilled.
    size_t zo = 0;
    asm volatile("" : "+s"(zo));
    const _Float16* pWth = WthF + zo;
    const _Float16* pWtx = WtxF + zo;
    const _Float16* pWfr = WfrF + zo;
    const _Float16* pWhr = WhrT + zo;
    const _Float16* pWwc = WwcT + zo;
    const _Float16* pWk  = WkT  + zo;
    const _Float16* pWr  = WrT  + zo;

    const float dInv = frcp(denomT[t]);

    // stage x, m, d for this timestep (f16 operand images)
    for (int i = tid * 8; i < tid * 8 + 8; ++i) {
      int row = i >> 7, col = i & 127;
      size_t g = ((size_t)(rb + row) * TT + t) * DD + col;
      xF[i] = (_Float16)values[g];
      mF[i] = (_Float16)masks[g];
      dF[i] = (_Float16)deltas[g];
    }
    __syncthreads();

    // GEMM1: gamma_h = exp(-relu(d @ Wth^T + bth)); fused decay h' = h * gamma_h
    {
      v16h a[4], b[4];
#pragma unroll
      for (int kc = 0; kc < 4; ++kc) {
        a[kc] = load_a16(dF, lane, kc * 32, DD);
        b[kc] = load_b16(pWth, lane, n0, kc * 32, DD);
      }
      v8f acc = {};
#pragma unroll
      for (int kc = 0; kc < 4; ++kc) WMMA_F16(acc, a[kc], b[kc]);
      const int col = n0 + cl;
      const float bias = bth[col];
#pragma unroll
      for (int r = 0; r < 8; ++r) {
        int row = rloc + r;
        float gh = fexp(-fmaxf(acc[r] + bias, 0.f));
        hB[row * HH + col] = (_Float16)((float)hA[row * HH + col] * gh);
      }
    }
    __syncthreads();

    // GEMM2: x_hat = h' @ Whr + bhr; fused x_c = m*x + (1-m)*x_hat
    {
      v16h a[4], b[4];
#pragma unroll
      for (int kc = 0; kc < 4; ++kc) {
        a[kc] = load_a16(hB, lane, kc * 32, HH);
        b[kc] = load_b16(pWhr, lane, n0, kc * 32, HH);
      }
      v8f acc = {};
#pragma unroll
      for (int kc = 0; kc < 4; ++kc) WMMA_F16(acc, a[kc], b[kc]);
      const int col = n0 + cl;
      const float bias = bhr[col];
#pragma unroll
      for (int r = 0; r < 8; ++r) {
        int row = rloc + r, idx = row * DD + col;
        float xh = acc[r] + bias;
        xhat32[idx] = xh;
        float m = (float)mF[idx], x = (float)xF[idx];
        xcF[idx] = (_Float16)(m * x + (1.f - m) * xh);
      }
    }
    __syncthreads();

    // GEMM3: z_hat = x_c @ Wfr^T + bfr ; GEMM4: gamma_x = exp(-relu(d @ Wtx^T + btx))
    {
      const int col = n0 + cl;
      {
        v16h a[4], b[4];
#pragma unroll
        for (int kc = 0; kc < 4; ++kc) {
          a[kc] = load_a16(xcF, lane, kc * 32, DD);
          b[kc] = load_b16(pWfr, lane, n0, kc * 32, DD);
        }
        v8f acc = {};
#pragma unroll
        for (int kc = 0; kc < 4; ++kc) WMMA_F16(acc, a[kc], b[kc]);
        float bias = bfr[col];
#pragma unroll
        for (int r = 0; r < 8; ++r) zhat32[(rloc + r) * DD + col] = acc[r] + bias;
      }
      {
        v16h a[4], b[4];
#pragma unroll
        for (int kc = 0; kc < 4; ++kc) {
          a[kc] = load_a16(dF, lane, kc * 32, DD);
          b[kc] = load_b16(pWtx, lane, n0, kc * 32, DD);
        }
        v8f acc = {};
#pragma unroll
        for (int kc = 0; kc < 4; ++kc) WMMA_F16(acc, a[kc], b[kc]);
        float bias = btx[col];
#pragma unroll
        for (int r = 0; r < 8; ++r)
          gxF[(rloc + r) * DD + col] = (_Float16)fexp(-fmaxf(acc[r] + bias, 0.f));
      }
    }
    __syncthreads();

    // GEMM5: beta = [gamma_x | m] @ Wwc + bwc; fused c_hat, c_c, imputation, loss
    {
      v8f acc = {};
      {
        v16h a[4], b[4];
#pragma unroll
        for (int kc = 0; kc < 4; ++kc) {
          a[kc] = load_a16(gxF, lane, kc * 32, DD);
          b[kc] = load_b16(pWwc, lane, n0, kc * 32, 2 * DD);
        }
#pragma unroll
        for (int kc = 0; kc < 4; ++kc) WMMA_F16(acc, a[kc], b[kc]);
      }
      {
        v16h a[4], b[4];
#pragma unroll
        for (int kc = 0; kc < 4; ++kc) {
          a[kc] = load_a16(mF, lane, kc * 32, DD);
          b[kc] = load_b16(pWwc, lane, n0, 128 + kc * 32, 2 * DD);
        }
#pragma unroll
        for (int kc = 0; kc < 4; ++kc) WMMA_F16(acc, a[kc], b[kc]);
      }
      const int col = n0 + cl;
      const float bias = bwc[col];
#pragma unroll
      for (int r = 0; r < 8; ++r) {
        int row = rloc + r, idx = row * DD + col;
        float beta = acc[r] + bias;
        float xh = xhat32[idx], zh = zhat32[idx];
        float ch = beta * zh + (1.f - beta) * xh;
        float m = (float)mF[idx], x = (float)xF[idx];
        float cc = m * x + (1.f - m) * ch;
        ccF[idx] = (_Float16)cc;
        outImp[((size_t)(rb + row) * TT + t) * DD + col] = cc;
        float l = (fabsf(x - xh) + fabsf(x - zh) + fabsf(x - ch)) * m * dInv;
        atomicAdd(&lossRow[row], l);
      }
    }
    __syncthreads();

    // GEMM6: gates = [c_c | m] @ Wk + h' @ Wr + bl; fused LSTM cell update
    {
      v8f acc[4] = {{}, {}, {}, {}};
#pragma unroll
      for (int kc = 0; kc < 8; ++kc) {
        v16h a = (kc < 4) ? load_a16(ccF, lane, kc * 32, DD)
                          : load_a16(mF, lane, (kc - 4) * 32, DD);
        v16h b0 = load_b16(pWk, lane, 0 * HH + n0, kc * 32, 2 * DD);
        v16h b1 = load_b16(pWk, lane, 1 * HH + n0, kc * 32, 2 * DD);
        v16h b2 = load_b16(pWk, lane, 2 * HH + n0, kc * 32, 2 * DD);
        v16h b3 = load_b16(pWk, lane, 3 * HH + n0, kc * 32, 2 * DD);
        WMMA_F16(acc[0], a, b0);
        WMMA_F16(acc[1], a, b1);
        WMMA_F16(acc[2], a, b2);
        WMMA_F16(acc[3], a, b3);
      }
#pragma unroll
      for (int kc = 0; kc < 4; ++kc) {
        v16h a = load_a16(hB, lane, kc * 32, HH);
        v16h b0 = load_b16(pWr, lane, 0 * HH + n0, kc * 32, HH);
        v16h b1 = load_b16(pWr, lane, 1 * HH + n0, kc * 32, HH);
        v16h b2 = load_b16(pWr, lane, 2 * HH + n0, kc * 32, HH);
        v16h b3 = load_b16(pWr, lane, 3 * HH + n0, kc * 32, HH);
        WMMA_F16(acc[0], a, b0);
        WMMA_F16(acc[1], a, b1);
        WMMA_F16(acc[2], a, b2);
        WMMA_F16(acc[3], a, b3);
      }
      const int col = n0 + cl;
      const float bi = bl[col], bf = bl[HH + col], bg = bl[2 * HH + col], bo2 = bl[3 * HH + col];
#pragma unroll
      for (int r = 0; r < 8; ++r) {
        int row = rloc + r, idx = row * HH + col;
        float ig = fsig(acc[0][r] + bi);
        float fg = fsig(acc[1][r] + bf);
        float gg = ftanh(acc[2][r] + bg);
        float og = fsig(acc[3][r] + bo2);
        float cN = fg * c32[idx] + ig * gg;
        c32[idx] = cN;
        hA[idx] = (_Float16)(og * ftanh(cN));    // new h for next step
      }
    }
    __syncthreads();
  }

  // predictions = relu(h @ Wd + bd) @ Wo + bo
  {
    int row = tid >> 4, j = tid & 15;
    float partial = 0.f;
    for (int nn = j; nn < DD; nn += 16) {
      float acc = bd[nn];
      for (int k = 0; k < HH; ++k)
        acc += (float)hA[row * HH + k] * (float)WdT[nn * HH + k];
      partial += fmaxf(acc, 0.f) * Wo[nn];
    }
    atomicAdd(&predRow[row], partial);
  }
  __syncthreads();
  if (tid < ROWS) {
    outPred[rb + tid] = predRow[tid] + bo[0];
    outLoss[rb + tid] = lossRow[tid] * (1.0f / (float)TT);
  }
}

// ---- host launch ----------------------------------------------------------
extern "C" void kernel_launch(void* const* d_in, const int* in_sizes, int n_in,
                              void* d_out, int out_size, void* d_ws, size_t ws_size,
                              hipStream_t stream) {
  const float* values = (const float*)d_in[0];
  const float* masks  = (const float*)d_in[1];
  const float* deltas = (const float*)d_in[2];
  const float* Wth = (const float*)d_in[3];  const float* bth = (const float*)d_in[4];
  const float* Wtx = (const float*)d_in[5];  const float* btx = (const float*)d_in[6];
  const float* Whr = (const float*)d_in[7];  const float* bhr = (const float*)d_in[8];
  const float* Wfr = (const float*)d_in[9];  const float* bfr = (const float*)d_in[10];
  const float* Wwc = (const float*)d_in[11]; const float* bwc = (const float*)d_in[12];
  const float* Wk  = (const float*)d_in[13];
  const float* Wr  = (const float*)d_in[14];
  const float* bl  = (const float*)d_in[15];
  const float* Wd  = (const float*)d_in[16]; const float* bd = (const float*)d_in[17];
  const float* Wo  = (const float*)d_in[18]; const float* bo = (const float*)d_in[19];

  // workspace layout (f16 weight images, then denom[t])
  _Float16* wsH  = (_Float16*)d_ws;
  _Float16* WthF = wsH + 0;        // [128][128] as-is (rows of Wth are B columns)
  _Float16* WtxF = wsH + 16384;    // [128][128] as-is
  _Float16* WfrF = wsH + 32768;    // [128][128] as-is
  _Float16* WhrT = wsH + 49152;    // [128][128] transposed
  _Float16* WwcT = wsH + 65536;    // [128][256] transposed
  _Float16* WkT  = wsH + 98304;    // [512][256] transposed
  _Float16* WrT  = wsH + 229376;   // [512][128] transposed
  _Float16* WdT  = wsH + 294912;   // [128][128] transposed
  float* denomT  = (float*)((char*)d_ws + (size_t)311296 * 2);

  cvt_f16<<<64, 256, 0, stream>>>(Wth, WthF, 16384);
  cvt_f16<<<64, 256, 0, stream>>>(Wtx, WtxF, 16384);
  cvt_f16<<<64, 256, 0, stream>>>(Wfr, WfrF, 16384);
  cvt_f16_T<<<64, 256, 0, stream>>>(Whr, WhrT, 128, 128);
  cvt_f16_T<<<128, 256, 0, stream>>>(Wwc, WwcT, 256, 128);
  cvt_f16_T<<<512, 256, 0, stream>>>(Wk, WkT, 256, 512);
  cvt_f16_T<<<256, 256, 0, stream>>>(Wr, WrT, 128, 512);
  cvt_f16_T<<<64, 256, 0, stream>>>(Wd, WdT, 128, 128);
  denom_kernel<<<TT, 256, 0, stream>>>(masks, denomT);

  float* out = (float*)d_out;
  float* outPred = out;                                  // [B,1]
  float* outImp  = out + BB;                             // [B,T,D]
  float* outLoss = out + BB + (size_t)BB * TT * DD;      // [B]

  rits_persistent<<<NBLK, 256, 0, stream>>>(
      values, masks, deltas,
      bth, btx, bhr, bfr, bwc, bl, bd, Wo, bo,
      WthF, WtxF, WfrF, WhrT, WwcT, WkT, WrT, WdT,
      denomT, outPred, outImp, outLoss);
}
